// SelfAttention_747324309955
// MI455X (gfx1250) — 16-bit and 8-bit matrix operand order repaired and checked by reading, compiled, not run on hardware
//
#include <hip/hip_runtime.h>
#include <stdint.h>

typedef __attribute__((ext_vector_type(8)))  int      v8i;
typedef __attribute__((ext_vector_type(8)))  float    v8f;
typedef __attribute__((ext_vector_type(16))) _Float16 v16h;

union V8I  { v8i v;  uint2 u2[4]; uint4 u4[2]; int e[8]; };
union V8F  { v8f v;  float e[8]; };
union V16H { v16h v; uint4 u4[2]; _Float16 e[16]; };

#define BQ   8
#define SQ   512
#define HQ   1024
#define NH   16
#define HD   64
#define INVS 0.125f

#define OFF_SX    ((size_t)0)
#define OFF_SW    ((size_t)4  << 20)
#define OFF_SCALE ((size_t)7  << 20)
#define OFF_QH    ((size_t)8  << 20)
#define OFF_S     ((size_t)32 << 20)
#define OFF_SVT   ((size_t)44 << 20)
#define OFF_P2    ((size_t)48 << 20)

#define OUT_CTX   ((size_t)0)
#define OUT_ATTN  ((size_t)4194304)
#define OUT_VAL   ((size_t)37748736)
#define OUT_ZERO  ((size_t)71303168)
#define OUT_QRY   ((size_t)71303169)
#define OUT_KEY   ((size_t)104857601)

__device__ __forceinline__ int8_t fsign8(float x) {
  return (x > 0.f) ? (int8_t)1 : ((x < 0.f) ? (int8_t)-1 : (int8_t)0);
}

__global__ void prep_weights(const float* __restrict__ wq, const float* __restrict__ wk,
                             const float* __restrict__ wv, int8_t* __restrict__ Sw,
                             float* __restrict__ scale) {
  int proj = blockIdx.x >> 10;
  int row  = blockIdx.x & 1023;
  const float* w    = (proj == 0) ? wq : (proj == 1) ? wk : wv;
  const float* wrow = w + (size_t)row * HQ;
  __shared__ float ssum[256];
  __shared__ float sabs[256];
  int tid = threadIdx.x;
  float s = 0.f, a = 0.f;
  for (int i = tid; i < HQ; i += 256) { float x = wrow[i]; s += x; a += fabsf(x); }
  ssum[tid] = s; sabs[tid] = a; __syncthreads();
  for (int off = 128; off > 0; off >>= 1) {
    if (tid < off) { ssum[tid] += ssum[tid + off]; sabs[tid] += sabs[tid + off]; }
    __syncthreads();
  }
  float mean = ssum[0] * (1.f / HQ);
  if (tid == 0) scale[proj * HQ + row] = sabs[0] * (1.f / HQ);
  int8_t* srow = Sw + ((size_t)proj * HQ + row) * HQ;
  for (int i = tid; i < HQ; i += 256) srow[i] = fsign8(wrow[i] - mean);
}

__global__ void binarize_x(const float* __restrict__ x, int8_t* __restrict__ Sx) {
  size_t base = ((size_t)blockIdx.x * 256 + threadIdx.x) * 4;
  float4 v = *(const float4*)(x + base);
  char4 o;
  o.x = fsign8(v.x); o.y = fsign8(v.y); o.z = fsign8(v.z); o.w = fsign8(v.w);
  *(char4*)(Sx + base) = o;
}

__global__ void qkv_gemm(const int8_t* __restrict__ Sx, const int8_t* __restrict__ Sw,
                         const float* __restrict__ scale,
                         const float* __restrict__ bq, const float* __restrict__ bk,
                         const float* __restrict__ bv,
                         _Float16* __restrict__ qh, int8_t* __restrict__ S,
                         int8_t* __restrict__ SvT) {
  int lane = threadIdx.x & 31, wvid = threadIdx.x >> 5;
  int flat = blockIdx.x * 8 + wvid;
  int p = flat >> 12, rem = flat & 4095;
  int mtile = rem >> 4, ngroup = rem & 15;
  int half = lane >> 4, l15 = lane & 15;
  int m = mtile * 16 + l15;
  const float* bias = (p == 0) ? bq : ((p == 1) ? bk : bv);
  const int8_t* swbase = Sw + (size_t)p * HQ * HQ;

  V8I acc[4];
#pragma unroll
  for (int j = 0; j < 4; ++j)
#pragma unroll
    for (int e = 0; e < 8; ++e) acc[j].e[e] = 0;

  for (int ks = 0; ks < 16; ++ks) {
    V8I A;
    const int8_t* ap = Sx + (size_t)m * HQ + ks * 64 + 16 * half;
    A.u2[0] = *(const uint2*)(ap +  0);
    A.u2[1] = *(const uint2*)(ap +  8);
    A.u2[2] = *(const uint2*)(ap + 32);
    A.u2[3] = *(const uint2*)(ap + 40);
#pragma unroll
    for (int j = 0; j < 4; ++j) {
      int n = (ngroup * 4 + j) * 16 + l15;
      const int8_t* bp = swbase + (size_t)n * HQ + ks * 64 + 16 * half;
      V8I Bv;
      Bv.u4[0] = *(const uint4*)(bp +  0);
      Bv.u4[1] = *(const uint4*)(bp + 32);
      acc[j].v = __builtin_amdgcn_wmma_i32_16x16x64_iu8(true, A.v, true, Bv.v,
                                                        acc[j].v, false, false);
    }
  }
#pragma unroll
  for (int j = 0; j < 4; ++j) {
    int col = (ngroup * 4 + j) * 16 + l15;
    float scl = scale[p * HQ + col];
    float bs  = bias[col];
#pragma unroll
    for (int r = 0; r < 8; ++r) {
      int row = mtile * 16 + r + 8 * half;
      float qv = scl * (float)acc[j].e[r] + bs;
      size_t idx = ((size_t)p << 22) + (size_t)row * HQ + col;
      qh[idx] = (_Float16)qv;
      int8_t sg = fsign8(qv);
      S[idx] = sg;
      if (p == 2) {
        int bb = row >> 9, s = row & 511, h = col >> 6, d = col & 63;
        SvT[((((size_t)bb * NH + h) * HD + d) << 9) + s] = sg;
      }
    }
  }
}

__global__ void gram(const _Float16* __restrict__ qh, float* __restrict__ out) {
  int lane = threadIdx.x & 31, wvid = threadIdx.x >> 5;
  int flat = blockIdx.x * 8 + wvid;
  int g = flat >> 15, rem = flat & 32767;
  int bh = rem >> 8, r2 = rem & 255;
  int s1t = r2 >> 3, s2g = r2 & 7;
  int half = lane >> 4, l15 = lane & 15;
  int b = bh >> 4, h = bh & 15;
  const _Float16* base = qh + ((size_t)g << 22) + (size_t)b * SQ * HQ + h * HD;
  int m = s1t * 16 + l15;

  V8F acc[4];
#pragma unroll
  for (int j = 0; j < 4; ++j)
#pragma unroll
    for (int e = 0; e < 8; ++e) acc[j].e[e] = 0.f;

#pragma unroll
  for (int kk = 0; kk < 2; ++kk) {
    int kb = kk * 32;
    V16H A;
    const _Float16* aptr = base + (size_t)m * HQ + kb + 8 * half;
    A.u4[0] = *(const uint4*)(aptr);
    A.u4[1] = *(const uint4*)(aptr + 16);
#pragma unroll
    for (int j = 0; j < 4; ++j) {
      int n = (s2g * 4 + j) * 16 + l15;
      const _Float16* bptr = base + (size_t)n * HQ + kb + 8 * half;
      V16H Bv;
      Bv.u4[0] = *(const uint4*)(bptr);
      Bv.u4[1] = *(const uint4*)(bptr + 16);
      acc[j].v = __builtin_amdgcn_wmma_f32_16x16x32_f16(false, A.v, false, Bv.v,
                                                        (short)0, acc[j].v, false, false);
    }
  }
  size_t gb = (g == 0) ? OUT_QRY : ((g == 1) ? OUT_KEY : OUT_VAL);
  float* o = out + gb + (size_t)bh * SQ * SQ;
#pragma unroll
  for (int j = 0; j < 4; ++j) {
    int col = (s2g * 4 + j) * 16 + l15;
#pragma unroll
    for (int r = 0; r < 8; ++r) {
      int s1 = s1t * 16 + r + 8 * half;
      o[(size_t)s1 * SQ + col] = INVS * acc[j].e[r];
    }
  }
}

__global__ void attn(const int8_t* __restrict__ Sq, const int8_t* __restrict__ Sk,
                     const float* __restrict__ mask, float* __restrict__ out,
                     int8_t* __restrict__ P2) {
  int lane = threadIdx.x & 31, wvid = threadIdx.x >> 5;
  int flat = blockIdx.x * 8 + wvid;
  int bh = flat >> 8, r2 = flat & 255;
  int s1t = r2 >> 3, s2g = r2 & 7;
  int half = lane >> 4, l15 = lane & 15;
  int b = bh >> 4, h = bh & 15;
  int m = s1t * 16 + l15;
  const int8_t* abase = Sq + (size_t)b * SQ * HQ + h * HD;
  const int8_t* bbase = Sk + (size_t)b * SQ * HQ + h * HD;

  V8I A;
  const int8_t* ap = abase + (size_t)m * HQ + 16 * half;
  A.u2[0] = *(const uint2*)(ap +  0);
  A.u2[1] = *(const uint2*)(ap +  8);
  A.u2[2] = *(const uint2*)(ap + 32);
  A.u2[3] = *(const uint2*)(ap + 40);

  V8I acc[4];
#pragma unroll
  for (int j = 0; j < 4; ++j)
#pragma unroll
    for (int e = 0; e < 8; ++e) acc[j].e[e] = 0;

#pragma unroll
  for (int j = 0; j < 4; ++j) {
    int n = (s2g * 4 + j) * 16 + l15;
    const int8_t* bp = bbase + (size_t)n * HQ + 16 * half;
    V8I Bv;
    Bv.u4[0] = *(const uint4*)(bp +  0);
    Bv.u4[1] = *(const uint4*)(bp + 32);
    acc[j].v = __builtin_amdgcn_wmma_i32_16x16x64_iu8(true, A.v, true, Bv.v,
                                                      acc[j].v, false, false);
  }
  float* o = out + OUT_ATTN + (size_t)bh * SQ * SQ;
  int8_t* p2 = P2 + (size_t)bh * SQ * SQ;
#pragma unroll
  for (int j = 0; j < 4; ++j) {
    int col = (s2g * 4 + j) * 16 + l15;
    float mk = mask[b * SQ + col];
#pragma unroll
    for (int r = 0; r < 8; ++r) {
      int s1 = s1t * 16 + r + 8 * half;
      float sc = (float)acc[j].e[r] * INVS + mk;
      o[(size_t)s1 * SQ + col] = sc;
      p2[(size_t)s1 * SQ + col] = (sc > 0.f) ? 2 : ((sc == 0.f) ? 1 : 0);
    }
  }
}

__global__ void ctx_gemm(const int8_t* __restrict__ P2, const int8_t* __restrict__ SvT,
                         float* __restrict__ out) {
  int lane = threadIdx.x & 31, wvid = threadIdx.x >> 5;
  int flat = blockIdx.x * 8 + wvid;
  int bh = flat >> 5, s1t = flat & 31;
  int half = lane >> 4, l15 = lane & 15;
  int b = bh >> 4, h = bh & 15;
  int m = s1t * 16 + l15;
  const int8_t* abase = P2 + (size_t)bh * SQ * SQ;
  const int8_t* bbase = SvT + (size_t)bh * HD * SQ;

  V8I acc[4];
#pragma unroll
  for (int j = 0; j < 4; ++j)
#pragma unroll
    for (int e = 0; e < 8; ++e) acc[j].e[e] = 0;

  for (int ks = 0; ks < 8; ++ks) {
    V8I A;
    const int8_t* ap = abase + (size_t)m * SQ + ks * 64 + 16 * half;
    A.u2[0] = *(const uint2*)(ap +  0);
    A.u2[1] = *(const uint2*)(ap +  8);
    A.u2[2] = *(const uint2*)(ap + 32);
    A.u2[3] = *(const uint2*)(ap + 40);
#pragma unroll
    for (int j = 0; j < 4; ++j) {
      int d = j * 16 + l15;
      const int8_t* bp = bbase + (size_t)d * SQ + ks * 64 + 16 * half;
      V8I Bv;
      Bv.u4[0] = *(const uint4*)(bp +  0);
      Bv.u4[1] = *(const uint4*)(bp + 32);
      acc[j].v = __builtin_amdgcn_wmma_i32_16x16x64_iu8(false, A.v, true, Bv.v,
                                                        acc[j].v, false, false);
    }
  }
#pragma unroll
  for (int j = 0; j < 4; ++j) {
    int col = h * HD + j * 16 + l15;
#pragma unroll
    for (int r = 0; r < 8; ++r) {
      int s1 = s1t * 16 + r + 8 * half;
      out[OUT_CTX + ((size_t)b * SQ + s1) * HQ + col] = 0.5f * (float)acc[j].e[r];
    }
  }
  if (blockIdx.x == 0 && threadIdx.x == 0) out[OUT_ZERO] = 0.f;
}

extern "C" void kernel_launch(void* const* d_in, const int* in_sizes, int n_in,
                              void* d_out, int out_size, void* d_ws, size_t ws_size,
                              hipStream_t stream) {
  const float* hs   = (const float*)d_in[0];
  const float* mask = (const float*)d_in[1];
  const float* wq   = (const float*)d_in[2];
  const float* bq   = (const float*)d_in[3];
  const float* wk   = (const float*)d_in[4];
  const float* bk   = (const float*)d_in[5];
  const float* wv   = (const float*)d_in[6];
  const float* bv   = (const float*)d_in[7];

  uint8_t* ws = (uint8_t*)d_ws;
  int8_t*   Sx    = (int8_t*)(ws + OFF_SX);
  int8_t*   Sw    = (int8_t*)(ws + OFF_SW);
  float*    scale = (float*)(ws + OFF_SCALE);
  _Float16* qh    = (_Float16*)(ws + OFF_QH);
  int8_t*   S     = (int8_t*)(ws + OFF_S);
  int8_t*   SvT   = (int8_t*)(ws + OFF_SVT);
  int8_t*   P2    = (int8_t*)(ws + OFF_P2);
  float*    out   = (float*)d_out;

  prep_weights<<<3072, 256, 0, stream>>>(wq, wk, wv, Sw, scale);
  binarize_x  <<<4096, 256, 0, stream>>>(hs, Sx);
  qkv_gemm    <<<1536, 256, 0, stream>>>(Sx, Sw, scale, bq, bk, bv, qh, S, SvT);
  gram        <<<12288, 256, 0, stream>>>(qh, out);
  attn        <<<4096, 256, 0, stream>>>(S, S + ((size_t)1 << 22), mask, out, P2);
  ctx_gemm    <<<512,  256, 0, stream>>>(P2, SvT, out);
}
